// MultiHeadAttention_32727650795939
// MI455X (gfx1250) — compile-verified, matrix operand order checked by reading, not run on hardware
//
#include <hip/hip_runtime.h>
#include <hip/hip_bf16.h>
#include <math.h>

typedef __attribute__((ext_vector_type(16))) _Float16 v16h;
typedef __attribute__((ext_vector_type(8)))  _Float16 v8h;
typedef __attribute__((ext_vector_type(4)))  _Float16 v4h;
typedef __attribute__((ext_vector_type(8)))  float    v8f;
typedef __attribute__((ext_vector_type(4)))  unsigned int u32x4;
typedef __attribute__((ext_vector_type(4)))  int          i32x4;
typedef __attribute__((ext_vector_type(8)))  int          i32x8;

#define IN_F   1024
#define NH     16
#define DIMD   64
#define BB     2
#define SS     2048
#define MTOT   (BB*SS)     // 4096 rows

#if defined(__has_builtin)
#if __has_builtin(__builtin_amdgcn_tensor_load_to_lds) && \
    __has_builtin(__builtin_amdgcn_s_wait_tensorcnt)
#define HAVE_TDM 1
#endif
#endif

// ---------------------------------------------------------------------------
// Fragment helpers.
// CDNA5 16-bit A-matrix 16x32 lane layout (ISA 7.12.2): lanes 0-15 hold
// K=0..7 / K=16..23; lanes 16-31 hold K=8..15 / K=24..31.  Each lane reads
// two contiguous 8-half (16B) chunks.  B stored pre-transposed ([N][K]) so B
// fragments use the identical pattern.
// ---------------------------------------------------------------------------
static __device__ __forceinline__ v16h frag_ld(const _Float16* base, int khalf) {
  v8h lo = *(const v8h*)(base + khalf * 8);
  v8h hi = *(const v8h*)(base + 16 + khalf * 8);
  v16h f;
#pragma unroll
  for (int i = 0; i < 8; ++i) { f[i] = lo[i]; f[i + 8] = hi[i]; }
  return f;
}

static __device__ __forceinline__ v8f wmma_f16(v16h a, v16h b, v8f c) {
  return __builtin_amdgcn_wmma_f32_16x16x32_f16(false, a, false, b,
                                                (short)0, c, false, false);
}

#ifdef HAVE_TDM
// ---------------------------------------------------------------------------
// TDM 2D tile load: global (tile_h rows x tile_w f16, row stride
// row_stride_elems f16) -> LDS at lds_off with hardware padding of 4 DWORDs
// (8 halves) every 32 DWORDs (64 halves), i.e. LDS row stride = 72 halves.
// D# layout per CDNA5 ISA (groups 0/1; 2D => later
// groups zero).  6-arg builtin form (clang-23 / therock-10.0 signature).
// ---------------------------------------------------------------------------
static __device__ __forceinline__ void tdm_load_2d_f16(
    const void* gptr, unsigned lds_off, unsigned tile_w, unsigned tile_h,
    unsigned row_stride_elems) {
  const unsigned long long ga = (unsigned long long)gptr;
  const unsigned TD0 = 1u << 20, TD1 = 1u << 20;   // huge tensor dims: no OOB
  u32x4 g0;
  g0[0] = 1u;                                     // count=1 (valid), user mode
  g0[1] = lds_off;                                // lds_addr (bytes)
  g0[2] = (unsigned)ga;                           // global_addr[31:0]
  g0[3] = (unsigned)((ga >> 32) & 0x01FFFFFFu) | (2u << 30); // [56:32] | type=2
  i32x8 g1;
  // w0: data_size=1 (2B) | pad_enable | pad_interval=4 (32 DW) | pad_amount=3 (4 DW)
  g1[0] = (int)((1u << 16) | (1u << 20) | (4u << 22) | (3u << 25));
  g1[1] = (int)((TD0 & 0xFFFFu) << 16);           // tensor_dim0[15:0]
  g1[2] = (int)((TD0 >> 16) | ((TD1 & 0xFFFFu) << 16));
  g1[3] = (int)((TD1 >> 16) | (tile_w << 16));    // tile_dim0
  g1[4] = (int)(tile_h);                          // tile_dim1 (tile_dim2=0)
  g1[5] = (int)row_stride_elems;                  // tensor_dim0_stride[31:0]
  g1[6] = 0; g1[7] = 0;                           // stride hi, dim1_stride
  i32x4 gz4; gz4[0] = 0; gz4[1] = 0; gz4[2] = 0; gz4[3] = 0;
  i32x8 gz8;
#pragma unroll
  for (int i = 0; i < 8; ++i) gz8[i] = 0;
  __builtin_amdgcn_tensor_load_to_lds(g0, g1, gz4, gz4, gz8, 0);
}
#endif

// ---------------------------------------------------------------------------
// f32 -> f16 elementwise convert (vectorized x4)
// ---------------------------------------------------------------------------
__global__ __launch_bounds__(256) void k_cvt(const float* __restrict__ in,
                                             _Float16* __restrict__ out, int n4) {
  int i = blockIdx.x * 256 + threadIdx.x;
  if (i < n4) {
    float4 v = ((const float4*)in)[i];
    v4h h;
    h[0] = (_Float16)v.x; h[1] = (_Float16)v.y;
    h[2] = (_Float16)v.z; h[3] = (_Float16)v.w;
    ((v4h*)out)[i] = h;
  }
}

// ---------------------------------------------------------------------------
// Weight transpose + convert:  in f32 [K][N]  ->  out f16 [N][K]
// ---------------------------------------------------------------------------
__global__ __launch_bounds__(256) void k_trans(const float* __restrict__ in,
                                               _Float16* __restrict__ out,
                                               int K, int N) {
  int k = blockIdx.x * 256 + threadIdx.x;
  int n = blockIdx.y;
  if (k < K) out[(size_t)n * K + k] = (_Float16)in[(size_t)k * N + n];
}

// ---------------------------------------------------------------------------
// Tiled WMMA GEMM:  C[M][N] = A[M][K] * B[K][N] + bias
//   128x128 block tile, BK=32, 256 threads = 8 waves, wave tile 32x64
//   (8 WMMAs / k-step / wave).  A optionally split at kSplit between A and A2
//   (virtual concat for the FC layer).  B pre-transposed: BT[N][K].
// ---------------------------------------------------------------------------
template <bool OUT16>
__global__ __launch_bounds__(256) void k_gemm(
    const _Float16* __restrict__ A, const _Float16* __restrict__ A2,
    const _Float16* __restrict__ BT, const float* __restrict__ bias,
    _Float16* __restrict__ o16, float* __restrict__ o32,
    int N, int K, int kSplit) {
  __shared__ __align__(16) _Float16 As[128 * 40];
  __shared__ __align__(16) _Float16 Bs[128 * 40];

  const int tid   = threadIdx.x;
  const int lane  = tid & 31;
  const int wave  = tid >> 5;
  const int wm    = wave & 3;   // 32-row slice
  const int wn    = wave >> 2;  // 64-col slice
  const int lq    = lane & 15;
  const int khalf = lane >> 4;
  const int m0    = blockIdx.x * 128;
  const int n0    = blockIdx.y * 128;
  const int ldr   = tid >> 2;        // 0..63
  const int ldc   = (tid & 3) * 8;   // 0,8,16,24

  v8f acc[2][4] = {};

  for (int k0 = 0; k0 < K; k0 += 32) {
    const _Float16* Aeff; int kk, Astr;
    if (k0 < kSplit) { Aeff = A;  kk = k0;          Astr = kSplit;     }
    else             { Aeff = A2; kk = k0 - kSplit; Astr = K - kSplit; }

    v8h av0 = *(const v8h*)(Aeff + (size_t)(m0 + ldr)      * Astr + kk + ldc);
    v8h av1 = *(const v8h*)(Aeff + (size_t)(m0 + 64 + ldr) * Astr + kk + ldc);
    v8h bv0 = *(const v8h*)(BT   + (size_t)(n0 + ldr)      * K    + k0 + ldc);
    v8h bv1 = *(const v8h*)(BT   + (size_t)(n0 + 64 + ldr) * K    + k0 + ldc);

    if (k0 + 32 < K) {   // hint next k-tile into cache (global_prefetch_b8)
      __builtin_prefetch(BT + (size_t)(n0 + ldr) * K + k0 + 32 + ldc, 0, 1);
      __builtin_prefetch(Aeff + (size_t)(m0 + ldr) * Astr + kk + 32 + ldc, 0, 1);
    }

    __syncthreads();
    *(v8h*)(&As[(ldr)      * 40 + ldc]) = av0;
    *(v8h*)(&As[(ldr + 64) * 40 + ldc]) = av1;
    *(v8h*)(&Bs[(ldr)      * 40 + ldc]) = bv0;
    *(v8h*)(&Bs[(ldr + 64) * 40 + ldc]) = bv1;
    __syncthreads();

    v16h af[2]; v16h bf[4];
#pragma unroll
    for (int i = 0; i < 2; ++i)
      af[i] = frag_ld(&As[(wm * 32 + i * 16 + lq) * 40], khalf);
#pragma unroll
    for (int j = 0; j < 4; ++j)
      bf[j] = frag_ld(&Bs[(wn * 64 + j * 16 + lq) * 40], khalf);
#pragma unroll
    for (int i = 0; i < 2; ++i)
#pragma unroll
      for (int j = 0; j < 4; ++j)
        acc[i][j] = wmma_f16(af[i], bf[j], acc[i][j]);
  }

  // C layout: VGPR r, lanes 0-15 -> M=r, N=lane; lanes 16-31 -> M=8+r.
#pragma unroll
  for (int j = 0; j < 4; ++j) {
    const int n  = n0 + wn * 64 + j * 16 + lq;
    const float bb = bias[n];
#pragma unroll
    for (int i = 0; i < 2; ++i) {
#pragma unroll
      for (int r = 0; r < 8; ++r) {
        const int m = m0 + wm * 32 + i * 16 + khalf * 8 + r;
        const float e = acc[i][j][r] + bb;
        if (OUT16) o16[(size_t)m * N + n] = (_Float16)e;
        else       o32[(size_t)m * N + n] = e;
      }
    }
  }
}

// ---------------------------------------------------------------------------
// Flash attention (keys == value projection, scale 1/8, per (b,h)):
//   grid (S/64, H, B), 128 threads = 4 waves; wave owns 16 query rows.
//   V tile staged by the Tensor Data Mover (hardware-padded rows) when
//   available; never materializes the 2048x2048 score matrix.
// ---------------------------------------------------------------------------
__global__ __launch_bounds__(128) void k_attn(const _Float16* __restrict__ q,
                                              const _Float16* __restrict__ v,
                                              _Float16* __restrict__ ctx) {
  __shared__ __align__(16) _Float16 Vs [64 * 72];   // [key][d]  (B^T for QV^T)
  __shared__ __align__(16) _Float16 VsT[64 * 72];   // [d][key]  (B^T for P*V)
  __shared__ __align__(16) _Float16 Ps [4][16 * 72];// per-wave exp(P) staging

  const int tid   = threadIdx.x;
  const int lane  = tid & 31;
  const int wave  = tid >> 5;
  const int lq    = lane & 15;
  const int khalf = lane >> 4;
  const int h     = blockIdx.y;
  const int b     = blockIdx.z;
  const int q0    = blockIdx.x * 64 + wave * 16;
  const float sc  = 0.125f;   // 1/sqrt(64)

  // Persistent A fragments for this wave's 16 query rows (K = d = 64).
  const _Float16* qp = q + ((size_t)(b * SS + q0 + lq) * NH + h) * DIMD;
  const v16h aq0 = frag_ld(qp,      khalf);
  const v16h aq1 = frag_ld(qp + 32, khalf);

  float mmax[8], lsum[8];
  v8f cacc[4] = {};
#pragma unroll
  for (int r = 0; r < 8; ++r) { mmax[r] = -1e30f; lsum[r] = 0.f; }

  for (int kt = 0; kt < SS / 64; ++kt) {
    __syncthreads();   // all waves done reading previous V tile
    const _Float16* vtile = v + ((size_t)(b * SS + kt * 64) * NH + h) * DIMD;
#ifdef HAVE_TDM
    if (wave == 0) {
      tdm_load_2d_f16(vtile, (unsigned)(size_t)(const void*)Vs,
                      64u, 64u, (unsigned)(NH * DIMD));
      __builtin_amdgcn_s_wait_tensorcnt(0);
    }
    __syncthreads();
    // Build transposed copy from the TDM-staged tile (LDS -> LDS).
#pragma unroll
    for (int c = 0; c < 4; ++c) {
      const int idx = c * 1024 + tid * 8;
      const int key = idx >> 6;
      const int d   = idx & 63;
      v8h vv = *(const v8h*)(&Vs[key * 72 + d]);
#pragma unroll
      for (int i = 0; i < 8; ++i) VsT[(d + i) * 72 + key] = vv[i];
    }
#else
    // Fallback: stage 64-key V tile into LDS, both orientations.
#pragma unroll
    for (int c = 0; c < 4; ++c) {
      const int idx = c * 1024 + tid * 8;
      const int key = idx >> 6;
      const int d   = idx & 63;
      v8h vv = *(const v8h*)(vtile + (size_t)key * NH * DIMD + d);
      *(v8h*)(&Vs[key * 72 + d]) = vv;
#pragma unroll
      for (int i = 0; i < 8; ++i) VsT[(d + i) * 72 + key] = vv[i];
    }
#endif
    __syncthreads();

    // Scores: 16 (q) x 64 (key) in four 16x16 tiles, K=64 -> 2 WMMAs each.
    v8f s[4];
#pragma unroll
    for (int t = 0; t < 4; ++t) {
      v8f z = {};
      v16h bf0 = frag_ld(&Vs[(t * 16 + lq) * 72], khalf);
      v16h bf1 = frag_ld(&Vs[(t * 16 + lq) * 72 + 32], khalf);
      z = wmma_f16(aq0, bf0, z);
      z = wmma_f16(aq1, bf1, z);
      s[t] = z;
    }

    // Online softmax (row stats live per-lane for rows r + 8*khalf).
    float alpha[8], rs[8];
#pragma unroll
    for (int r = 0; r < 8; ++r) {
      float mx = fmaxf(fmaxf(s[0][r], s[1][r]), fmaxf(s[2][r], s[3][r]));
#pragma unroll
      for (int o = 1; o < 16; o <<= 1) mx = fmaxf(mx, __shfl_xor(mx, o, 32));
      const float mn = fmaxf(mmax[r], mx * sc);
      alpha[r] = __expf(mmax[r] - mn);
      mmax[r]  = mn;
      rs[r]    = 0.f;
    }
#pragma unroll
    for (int t = 0; t < 4; ++t) {
#pragma unroll
      for (int r = 0; r < 8; ++r) {
        const float p = __expf(s[t][r] * sc - mmax[r]);
        rs[r] += p;
        Ps[wave][(r + khalf * 8) * 72 + t * 16 + lq] = (_Float16)p;
      }
    }
#pragma unroll
    for (int r = 0; r < 8; ++r) {
      float t = rs[r];
#pragma unroll
      for (int o = 1; o < 16; o <<= 1) t += __shfl_xor(t, o, 32);
      lsum[r] = lsum[r] * alpha[r] + t;
    }
#pragma unroll
    for (int t = 0; t < 4; ++t)
#pragma unroll
      for (int r = 0; r < 8; ++r) cacc[t][r] *= alpha[r];

    // Context accumulate: [16 x 64keys] * [64keys x 64d].
    const v16h p0 = frag_ld(&Ps[wave][lq * 72], khalf);
    const v16h p1 = frag_ld(&Ps[wave][lq * 72 + 32], khalf);
#pragma unroll
    for (int t = 0; t < 4; ++t) {
      v16h bv0 = frag_ld(&VsT[(t * 16 + lq) * 72], khalf);
      v16h bv1 = frag_ld(&VsT[(t * 16 + lq) * 72 + 32], khalf);
      cacc[t] = wmma_f16(p0, bv0, cacc[t]);
      cacc[t] = wmma_f16(p1, bv1, cacc[t]);
    }
  }

  // Normalize and emit context f16 [b][s][h][d].
#pragma unroll
  for (int r = 0; r < 8; ++r) {
    const float inv = 1.0f / lsum[r];
    const int m = q0 + khalf * 8 + r;
#pragma unroll
    for (int t = 0; t < 4; ++t)
      ctx[((size_t)(b * SS + m) * NH + h) * DIMD + t * 16 + lq] =
          (_Float16)(cacc[t][r] * inv);
  }
}

// ---------------------------------------------------------------------------
// LayerNorm over last dim (1024), in place on the FC output.
// ---------------------------------------------------------------------------
__global__ __launch_bounds__(256) void k_ln(float* __restrict__ x,
                                            const float* __restrict__ g,
                                            const float* __restrict__ be) {
  __shared__ float rA[8], rB[8];
  const int tid = threadIdx.x, lane = tid & 31, wave = tid >> 5;
  float* row = x + (size_t)blockIdx.x * IN_F;
  float vv[4], s = 0.f, s2 = 0.f;
#pragma unroll
  for (int i = 0; i < 4; ++i) {
    float t = row[tid + i * 256];
    vv[i] = t; s += t; s2 += t * t;
  }
#pragma unroll
  for (int o = 1; o < 32; o <<= 1) { s += __shfl_xor(s, o, 32); s2 += __shfl_xor(s2, o, 32); }
  if (lane == 0) { rA[wave] = s; rB[wave] = s2; }
  __syncthreads();
  float ts = 0.f, ts2 = 0.f;
#pragma unroll
  for (int i = 0; i < 8; ++i) { ts += rA[i]; ts2 += rB[i]; }
  const float mean = ts * (1.f / IN_F);
  const float var  = ts2 * (1.f / IN_F) - mean * mean;
  const float rstd = rsqrtf(var + 1e-5f);
#pragma unroll
  for (int i = 0; i < 4; ++i) {
    const int c = tid + i * 256;
    row[c] = (vv[i] - mean) * rstd * g[c] + be[c];
  }
}

// ---------------------------------------------------------------------------
extern "C" void kernel_launch(void* const* d_in, const int* in_sizes, int n_in,
                              void* d_out, int out_size, void* d_ws, size_t ws_size,
                              hipStream_t stream) {
  (void)in_sizes; (void)n_in; (void)out_size; (void)ws_size;
  const float* query = (const float*)d_in[0];
  const float* value = (const float*)d_in[1];
  const float* Wq    = (const float*)d_in[2];
  const float* bq    = (const float*)d_in[3];
  const float* Wv    = (const float*)d_in[4];
  const float* bv    = (const float*)d_in[5];
  const float* Wfc   = (const float*)d_in[6];
  const float* bfc   = (const float*)d_in[7];
  const float* gamma = (const float*)d_in[8];
  const float* beta  = (const float*)d_in[9];
  float* out = (float*)d_out;

  char* ws = (char*)d_ws;
  const size_t szAct = (size_t)MTOT * IN_F * sizeof(_Float16);   // 8 MB
  const size_t szW   = (size_t)IN_F * IN_F * sizeof(_Float16);   // 2 MB
  _Float16* qry16 = (_Float16*)ws;                 ws += szAct;
  _Float16* val16 = (_Float16*)ws;                 ws += szAct;
  _Float16* q16   = (_Float16*)ws;                 ws += szAct;
  _Float16* v16   = (_Float16*)ws;                 ws += szAct;
  _Float16* ctx16 = (_Float16*)ws;                 ws += szAct;
  _Float16* WqT   = (_Float16*)ws;                 ws += szW;
  _Float16* WvT   = (_Float16*)ws;                 ws += szW;
  _Float16* WfcT  = (_Float16*)ws;                 ws += 2 * szW;

  // 1) dtype conversion / weight transposes
  const int n4 = MTOT * IN_F / 4;
  k_cvt<<<dim3((n4 + 255) / 256), dim3(256), 0, stream>>>(query, qry16, n4);
  k_cvt<<<dim3((n4 + 255) / 256), dim3(256), 0, stream>>>(value, val16, n4);
  k_trans<<<dim3(IN_F / 256, IN_F), dim3(256), 0, stream>>>(Wq, WqT, IN_F, IN_F);
  k_trans<<<dim3(IN_F / 256, IN_F), dim3(256), 0, stream>>>(Wv, WvT, IN_F, IN_F);
  k_trans<<<dim3((2 * IN_F) / 256, IN_F), dim3(256), 0, stream>>>(Wfc, WfcT, 2 * IN_F, IN_F);

  // 2) Q / V projections (f16 WMMA GEMM, fused bias)
  dim3 gg(MTOT / 128, IN_F / 128);
  k_gemm<true><<<gg, dim3(256), 0, stream>>>(qry16, qry16, WqT, bq, q16, nullptr,
                                             IN_F, IN_F, IN_F);
  k_gemm<true><<<gg, dim3(256), 0, stream>>>(val16, val16, WvT, bv, v16, nullptr,
                                             IN_F, IN_F, IN_F);

  // 3) flash attention (TDM-staged V tiles)
  k_attn<<<dim3(SS / 64, NH, BB), dim3(128), 0, stream>>>(q16, v16, ctx16);

  // 4) FC over virtual concat([context, query]) with fused bias -> d_out (f32)
  k_gemm<false><<<gg, dim3(256), 0, stream>>>(ctx16, qry16, WfcT, bfc, nullptr, out,
                                              IN_F, 2 * IN_F, IN_F);

  // 5) LayerNorm in place
  k_ln<<<dim3(MTOT), dim3(256), 0, stream>>>(out, gamma, beta);
}
